// Attention25d_86775519248925
// MI455X (gfx1250) — matrix operand order repaired and checked by reading, not compiled, not run on hardware
//
#include <hip/hip_runtime.h>

// ---------------------------------------------------------------------------
// Windowed 2.5D attention (Swin-style), MI455X / gfx1250, wave32 + WMMA bf16.
//   x:(8192,49,128) f32 -> qkv -> per-head softmax(QK^T+bias) V -> out proj
// All four GEMMs run on v_wmma_f32_16x16x32_bf16.
// Tile scheduling is wave-uniform (readfirstlane) so q/k/v epilogue selection
// is scalar control flow instead of EXEC-masked lane branches.
// ---------------------------------------------------------------------------

typedef __bf16 bf16_t;
typedef __attribute__((ext_vector_type(16))) bf16_t v16bf;
typedef __attribute__((ext_vector_type(8)))  bf16_t v8bf;
typedef __attribute__((ext_vector_type(8)))  float  v8f;

#define N_TOK   49
#define NP      64      // padded token count
#define DIM     128
#define HEADS   4
#define DHEAD   32
#define QKV_N   384
#define SCALE_F 0.17677669529663687f   // 32^-0.5

struct __attribute__((aligned(8))) US4 { unsigned short a, b, c, d; };

__device__ __forceinline__ unsigned short f2bf(float f) {
    unsigned int u = __float_as_uint(f);
    unsigned int r = u + 0x7fffu + ((u >> 16) & 1u);   // round-to-nearest-even
    return (unsigned short)(r >> 16);
}

// A fragment: 16x32 bf16 tile, contiguous K order (matches load_frag_b):
//  lane<16: K = 0..15  (row M = lane)
//  lane>=16: K = 16..31 (row M = lane-16)
__device__ __forceinline__ v16bf load_frag_a(const unsigned short* base, int stride,
                                             int r0, int k0, int lane) {
    int m    = r0 + (lane & 15);
    int half = lane >> 4;
    const v8bf* plo = (const v8bf*)(base + m * stride + k0 + half * 16);
    const v8bf* phi = (const v8bf*)(base + m * stride + k0 + half * 16 + 8);
    v8bf lo = *plo;
    v8bf hi = *phi;
    v16bf out;
#pragma unroll
    for (int i = 0; i < 8; ++i) { out[i] = lo[i]; out[8 + i] = hi[i]; }
    return out;
}

// B fragment: 32x16 bf16 tile read from an [N][K]-major array (so each lane's
// 16 K-values are contiguous):  lane<16 -> K = k0..k0+15, lane>=16 -> +16,
// column N = n0 + (lane&15).
template <typename PTR>
__device__ __forceinline__ v16bf load_frag_b(PTR base, int stride,
                                             int n0, int k0, int lane) {
    int n  = n0 + (lane & 15);
    int kb = k0 + (lane >> 4) * 16;
    return *(const v16bf*)(base + n * stride + kb);
}

__device__ __forceinline__ v8f wmma_bf16(v16bf a, v16bf b, v8f c) {
    return __builtin_amdgcn_wmma_f32_16x16x32_bf16(false, a, false, b,
                                                   (short)0, c, false, false);
}

// ---------------------------------------------------------------------------
// Prep kernel: convert + transpose weights to bf16 [N][K], expand bias table
//   wqkvT : [384][128] bf16       (98304 B)
//   woutT : [128][128] bf16       (32768 B)
//   biasx : [4][49][64] f32       (50176 B)   biasx[h][i][j] = table[idx(i,j)][h]
// ---------------------------------------------------------------------------
__global__ void attn25d_prep(const float* __restrict__ w_qkv,
                             const float* __restrict__ w_out,
                             const float* __restrict__ bias_table,
                             unsigned short* __restrict__ wqkvT,
                             unsigned short* __restrict__ woutT,
                             float* __restrict__ biasx) {
    int i = blockIdx.x * 256 + threadIdx.x;
    if (i < QKV_N * DIM) {
        int n = i >> 7, k = i & 127;
        wqkvT[n * DIM + k] = f2bf(w_qkv[k * QKV_N + n]);
    }
    int j = i - QKV_N * DIM;
    if (j >= 0 && j < DIM * DIM) {
        int n = j >> 7, k = j & 127;
        woutT[n * DIM + k] = f2bf(w_out[k * DIM + n]);
    }
    int m = i - (QKV_N * DIM + DIM * DIM);
    if (m >= 0 && m < HEADS * N_TOK * NP) {
        int h = m / (N_TOK * NP);
        int r = m % (N_TOK * NP);
        int ii = r >> 6, jj = r & 63;
        float v = 0.f;
        if (jj < N_TOK) {
            int d0 = (ii / 7) - (jj / 7) + 6;
            int d1 = (ii % 7) - (jj % 7) + 6;
            v = bias_table[(d0 * 13 + d1) * HEADS + h];
        }
        biasx[m] = v;
    }
}

// ---------------------------------------------------------------------------
// Main kernel: one workgroup (8 wave32) per window.
// Dynamic LDS (131072 B):
//   xs   : bf16 [64][128]   (16 KB)  x staging; reused as attn@V output
//   qs   : bf16 [64][128]   (16 KB)  pre-scaled Q
//   ks   : bf16 [64][128]   (16 KB)
//   vT   : bf16 [128][64]   (16 KB)  V transposed (dim-major)
//   simf : f32  [4][64][64] (64 KB)  scores; probs rewritten in place as bf16
// ---------------------------------------------------------------------------
__global__ void __launch_bounds__(256)
attn25d_main(const float* __restrict__ xg_all,
             const unsigned short* __restrict__ wqkvT,
             const unsigned short* __restrict__ woutT,
             const float* __restrict__ biasx,
             float* __restrict__ outg) {
    extern __shared__ unsigned char smem[];
    unsigned short* xs   = (unsigned short*)smem;          // 8192 us
    unsigned short* qs   = xs + NP * DIM;                  // 8192 us
    unsigned short* ks   = qs + NP * DIM;                  // 8192 us
    unsigned short* vT   = ks + NP * DIM;                  // 8192 us
    float*          simf = (float*)(vT + DIM * NP);        // 16384 f32

    const int tid  = threadIdx.x;
    // wave index is uniform across the wave: pin it to an SGPR so all tile
    // scheduling (t, mt, nt, sec, h) becomes scalar control flow.
    const int wave = __builtin_amdgcn_readfirstlane(tid >> 5);
    const int lane = tid & 31;
    const int win  = blockIdx.x;

    const float* xg = xg_all + (size_t)win * N_TOK * DIM;

    // ---- stage x -> LDS bf16, zero-pad rows 49..63 (float4 loads) ----------
    {
        const float4* xg4 = (const float4*)xg;
        for (int e = tid; e < NP * (DIM / 4); e += 256) {
            int row = e >> 5;            // DIM/4 = 32 float4 per row
            int c4  = e & 31;
            US4 p = {0, 0, 0, 0};
            if (row < N_TOK) {
                float4 f = xg4[row * 32 + c4];
                p.a = f2bf(f.x); p.b = f2bf(f.y); p.c = f2bf(f.z); p.d = f2bf(f.w);
            }
            *(US4*)(xs + row * DIM + c4 * 4) = p;
        }
    }
    __syncthreads();

    // ---- QKV GEMM: (64x128) @ (128x384) = 96 tiles of 16x16 ----------------
    for (int t = wave; t < 4 * 24; t += 8) {
        int mt = t & 3;      // M tile 0..3
        int nt = t >> 2;     // N tile 0..23 (scalar)
        v8f acc = {0.f, 0.f, 0.f, 0.f, 0.f, 0.f, 0.f, 0.f};
#pragma unroll
        for (int k0 = 0; k0 < DIM; k0 += 32) {
            v16bf a = load_frag_a(xs, DIM, mt * 16, k0, lane);
            v16bf b = load_frag_b(wqkvT, DIM, nt * 16, k0, lane);
            acc = wmma_bf16(a, b, acc);
        }
        int sec    = nt >> 3;                             // 0=q 1=k 2=v (scalar)
        int col_in = (nt - sec * 8) * 16 + (lane & 15);   // 0..127
        int mbase  = mt * 16 + ((lane >> 4) << 3);
        if (sec == 0) {
#pragma unroll
            for (int v = 0; v < 8; ++v)
                qs[(mbase + v) * DIM + col_in] = f2bf(acc[v] * SCALE_F);
        } else if (sec == 1) {
#pragma unroll
            for (int v = 0; v < 8; ++v)
                ks[(mbase + v) * DIM + col_in] = f2bf(acc[v]);
        } else {
#pragma unroll
            for (int v = 0; v < 8; ++v)
                vT[col_in * NP + (mbase + v)] = f2bf(acc[v]);
        }
    }
    __syncthreads();

    // ---- scores: per head (64x32) @ (32x64) -> simf f32 --------------------
    for (int t = wave; t < HEADS * 16; t += 8) {
        int h   = t >> 4;
        int rem = t & 15;
        int mt  = rem >> 2;
        int nt  = rem & 3;
        v8f acc = {0.f, 0.f, 0.f, 0.f, 0.f, 0.f, 0.f, 0.f};
        v16bf a = load_frag_a(qs, DIM, mt * 16, h * DHEAD, lane);
        v16bf b = load_frag_b(ks, DIM, nt * 16, h * DHEAD, lane); // K^T via [tok][dim]
        acc = wmma_bf16(a, b, acc);
        float* dst = simf + h * (NP * NP);
        int n = nt * 16 + (lane & 15);
        int mbase = mt * 16 + ((lane >> 4) << 3);
#pragma unroll
        for (int v = 0; v < 8; ++v) dst[(mbase + v) * NP + n] = acc[v];
    }
    __syncthreads();

    // ---- softmax + bias: 256 threads = 4 heads x 64 rows -------------------
    {
        int h = tid >> 6, i = tid & 63;
        float* row = simf + h * (NP * NP) + i * NP;
        unsigned short* prow = (unsigned short*)row;   // probs in place (bf16)
        if (i < N_TOK) {
            const float* brow = biasx + (h * N_TOK + i) * NP;
            float mx = -3.0e38f;
            for (int j = 0; j < N_TOK; ++j) {
                float s = row[j] + brow[j];
                row[j] = s;
                mx = fmaxf(mx, s);
            }
            float sum = 0.f;
            for (int j = 0; j < N_TOK; ++j) {
                float e = __expf(row[j] - mx);
                row[j] = e;
                sum += e;
            }
            float inv = 1.f / sum;
            for (int j = 0; j < N_TOK; ++j) prow[j] = f2bf(row[j] * inv);
            for (int j = N_TOK; j < NP; ++j) prow[j] = 0;
        } else {
            for (int j = 0; j < NP; ++j) prow[j] = 0;
        }
    }
    __syncthreads();

    // ---- attn @ V: per head (64x64) @ (64x32) -> xs (reused, bf16) ---------
    for (int t = wave; t < HEADS * 8; t += 8) {
        int h   = t >> 3;
        int rem = t & 7;
        int mt  = rem >> 1;
        int nt  = rem & 1;
        const unsigned short* ps = (const unsigned short*)(simf + h * (NP * NP));
        v8f acc = {0.f, 0.f, 0.f, 0.f, 0.f, 0.f, 0.f, 0.f};
#pragma unroll
        for (int k0 = 0; k0 < NP; k0 += 32) {
            v16bf a = load_frag_a(ps, 2 * NP, mt * 16, k0, lane); // f32 row = 128 us stride
            v16bf b = load_frag_b(vT, NP, h * DHEAD + nt * 16, k0, lane);
            acc = wmma_bf16(a, b, acc);
        }
        int col   = h * DHEAD + nt * 16 + (lane & 15);
        int mbase = mt * 16 + ((lane >> 4) << 3);
#pragma unroll
        for (int v = 0; v < 8; ++v) xs[(mbase + v) * DIM + col] = f2bf(acc[v]);
    }
    __syncthreads();

    // ---- out projection: (64x128) @ (128x128), f32 straight to HBM ---------
    for (int t = wave; t < 4 * 8; t += 8) {
        int mt = t & 3;
        int nt = t >> 2;
        v8f acc = {0.f, 0.f, 0.f, 0.f, 0.f, 0.f, 0.f, 0.f};
#pragma unroll
        for (int k0 = 0; k0 < DIM; k0 += 32) {
            v16bf a = load_frag_a(xs, DIM, mt * 16, k0, lane);
            v16bf b = load_frag_b(woutT, DIM, nt * 16, k0, lane);
            acc = wmma_bf16(a, b, acc);
        }
        int n     = nt * 16 + (lane & 15);
        int mbase = mt * 16 + ((lane >> 4) << 3);
#pragma unroll
        for (int v = 0; v < 8; ++v) {
            int m = mbase + v;
            if (m < N_TOK)
                outg[((size_t)win * N_TOK + m) * DIM + n] = acc[v];
        }
    }
}

// ---------------------------------------------------------------------------
extern "C" void kernel_launch(void* const* d_in, const int* in_sizes, int n_in,
                              void* d_out, int out_size, void* d_ws, size_t ws_size,
                              hipStream_t stream) {
    const float* x          = (const float*)d_in[0];
    const float* w_qkv      = (const float*)d_in[1];
    const float* w_out      = (const float*)d_in[2];
    const float* bias_table = (const float*)d_in[3];

    // workspace carve (181248 B total)
    unsigned short* wqkvT = (unsigned short*)d_ws;                         // 98304 B
    unsigned short* woutT = (unsigned short*)((char*)d_ws + 98304);        // 32768 B
    float*          biasx = (float*)((char*)d_ws + 131072);                // 50176 B

    {
        int total  = QKV_N * DIM + DIM * DIM + HEADS * N_TOK * NP;  // 78080
        int blocks = (total + 255) / 256;
        attn25d_prep<<<blocks, 256, 0, stream>>>(w_qkv, w_out, bias_table,
                                                 wqkvT, woutT, biasx);
    }

    const int nwin = 4 * 16 * 16 * 8;           // 8192 windows
    const size_t lds_bytes = (size_t)(4 * NP * DIM * 2 + HEADS * NP * NP * 4); // 131072
    attn25d_main<<<nwin, 256, lds_bytes, stream>>>(x, wqkvT, woutT, biasx,
                                                   (float*)d_out);
}
